// CustomAttn_26663156974089
// MI455X (gfx1250) — compile-verified, matrix operand order checked by reading, not run on hardware
//
#include <hip/hip_runtime.h>
#include <hip/hip_bf16.h>
#include <stdint.h>

typedef __attribute__((ext_vector_type(16))) _Float16 v16h;
typedef __attribute__((ext_vector_type(8)))  _Float16 h8;
typedef __attribute__((ext_vector_type(8)))  float    v8f;

union F16x16 { v16h v; h8 h[2]; };

#define WMMA_F16(a, b, c) \
    __builtin_amdgcn_wmma_f32_16x16x32_f16(false, (a), false, (b), (short)0, (c), false, false)

// Async global->LDS copy of 16 bytes (CDNA5 GLOBAL_LOAD_ASYNC_TO_LDS_B128,
// tracked by ASYNCcnt). ldsaddr = low 32 bits of the generic LDS pointer
// (aperture spec: generic addr [31:0] is the LDS byte address).
__device__ __forceinline__ void async_ld_lds_b128(uint32_t ldsaddr, const void* gaddr) {
    asm volatile("global_load_async_to_lds_b128 %0, %1, off"
                 :: "v"(ldsaddr), "v"((uint64_t)(uintptr_t)gaddr)
                 : "memory");
}
__device__ __forceinline__ void wait_asynccnt0() {
    asm volatile("s_wait_asynccnt 0x0" ::: "memory");
}

// ---------------------------------------------------------------------------
// f32 -> f16 conversion
// ---------------------------------------------------------------------------
__global__ __launch_bounds__(256) void cvt_f32_f16_kernel(const float* __restrict__ src,
                                                          _Float16* __restrict__ dst, int n) {
    int i = blockIdx.x * 256 + threadIdx.x;
    if (i < n) dst[i] = (_Float16)src[i];
}

// ---------------------------------------------------------------------------
// C[M,N] = A[M,K] * B[N,K]^T   (both operands row-major, K contiguous)
// block = 128 threads = 4 waves; block tile 64x128, wave tile 16x128
// (8 f32 accumulators = 64 VGPRs). The 128x32-half B tile is staged
// cooperatively in LDS via global_load_async_to_lds_b128, double-buffered:
// tile i+1's async DMA overlaps tile i's 8 WMMAs per wave. A rows are unique
// per wave and stay on the direct global_load_b128 path (one A fragment now
// feeds 8 WMMAs). B fragments are preloaded 4-at-a-time so each WMMA burst
// pays a single s_wait_dscnt.
// ---------------------------------------------------------------------------
__global__ __launch_bounds__(128) void gemm_rt_f16_kernel(const _Float16* __restrict__ A,
                                                          const _Float16* __restrict__ B,
                                                          float* __restrict__ C,
                                                          int M, int N, int K) {
    __shared__ _Float16 lds_b[2][128][32];   // 2 x 8KB double buffer

    const int tid  = threadIdx.x;
    const int wave = tid >> 5;
    const int lane = tid & 31;
    const int hi   = lane >> 4;      // 0: lanes 0-15, 1: lanes 16-31
    const int l16  = lane & 15;

    const int row0 = blockIdx.y * 64 + wave * 16;
    const int col0 = blockIdx.x * 128;

    // Cooperative async stage of B[col0..col0+127][k0..k0+31] into lds_b[buf].
    // 512 16-byte chunks, 4 per thread. chunk c: row = c>>2, part = c&3.
    auto stage_b = [&](int k0, int buf) {
#pragma unroll
        for (int j = 0; j < 4; ++j) {
            const int c    = tid + j * 128;
            const int rowb = c >> 2;
            const int part = (c & 3) * 8;
            const _Float16* g = B + (size_t)(col0 + rowb) * K + k0 + part;
            async_ld_lds_b128((uint32_t)(uintptr_t)&lds_b[buf][rowb][part], g);
        }
    };

    v8f acc[8] = {};
    const _Float16* arow = A + (size_t)(row0 + l16) * K;

    stage_b(0, 0);

    for (int k0 = 0; k0 < K; k0 += 32) {
        const int buf = (k0 >> 5) & 1;

        wait_asynccnt0();      // tile k0 resident in lds_b[buf]
        __syncthreads();       // visible to all waves; all waves done with buf^1

        if (k0 + 32 < K) stage_b(k0 + 32, buf ^ 1);   // DMA overlaps WMMAs below

        const int koff = k0 + hi * 8;
        F16x16 a;
        a.h[0] = *(const h8*)(arow + koff);
        a.h[1] = *(const h8*)(arow + koff + 16);

#pragma unroll
        for (int g = 0; g < 2; ++g) {
            F16x16 b[4];
#pragma unroll
            for (int t = 0; t < 4; ++t) {
                const int rowb = g * 64 + t * 16 + l16;
                b[t].h[0] = *(const h8*)&lds_b[buf][rowb][hi * 8];
                b[t].h[1] = *(const h8*)&lds_b[buf][rowb][hi * 8 + 16];
            }
#pragma unroll
            for (int t = 0; t < 4; ++t)
                acc[g * 4 + t] = WMMA_F16(a.v, b[t].v, acc[g * 4 + t]);
        }
    }

#pragma unroll
    for (int t = 0; t < 8; ++t) {
#pragma unroll
        for (int i = 0; i < 8; ++i) {
            const int r = row0 + i + 8 * hi;
            const int c = col0 + t * 16 + l16;
            C[(size_t)r * N + c] = acc[t][i];
        }
    }
}

// ---------------------------------------------------------------------------
// RoPE + layout shuffle.
// qkv  : [S, 3072] f32  (q | k | v, each [S, H*64])
// q_h  : [H, S, 64] f16 (rotated)
// k_h  : [H, S, 64] f16 (rotated)
// v_t  : [H, 64, S] f16 (transposed for contiguous-K PV fragments)
// ---------------------------------------------------------------------------
__global__ __launch_bounds__(256) void rope_kernel(const float* __restrict__ qkv,
                                                   _Float16* __restrict__ q_h,
                                                   _Float16* __restrict__ k_h,
                                                   _Float16* __restrict__ v_t) {
    constexpr int S = 2048, HD = 64;
    const int idx = blockIdx.x * 256 + threadIdx.x;   // S*H*32 threads
    const int s = idx >> 9;
    const int r = idx & 511;
    const int h = r >> 5;
    const int d = r & 31;

    const float inv = __powf(10000.0f, -(float)(2 * d) / 64.0f);
    const float f = (float)s * inv;
    float sn, cs;
    __sincosf(f, &sn, &cs);

    const float* base = qkv + (size_t)s * 3072;
    const int e = h * HD + d;

    const float q1 = base[e],        q2 = base[e + 32];
    const float k1 = base[1024 + e], k2 = base[1024 + e + 32];
    const float v1 = base[2048 + e], v2 = base[2048 + e + 32];

    _Float16* qr = q_h + ((size_t)h * S + s) * HD;
    _Float16* kr = k_h + ((size_t)h * S + s) * HD;
    qr[d]      = (_Float16)(q1 * cs - q2 * sn);
    qr[d + 32] = (_Float16)(q2 * cs + q1 * sn);
    kr[d]      = (_Float16)(k1 * cs - k2 * sn);
    kr[d + 32] = (_Float16)(k2 * cs + k1 * sn);

    v_t[((size_t)h * HD + d) * S + s]      = (_Float16)v1;
    v_t[((size_t)h * HD + d + 32) * S + s] = (_Float16)v2;
}

// ---------------------------------------------------------------------------
// Fused flash attention (causal), WMMA for QK^T and PV.
// grid = (S/64, H), block = 128 (4 independent waves, one 16-query tile each).
// Per 32-key block: 4 wmma (scores) + 4 wmma (PV). Scores never hit HBM.
// ---------------------------------------------------------------------------
__device__ __forceinline__ float half_max16(float x) {
#pragma unroll
    for (int m = 1; m < 16; m <<= 1) x = fmaxf(x, __shfl_xor(x, m, 32));
    return x;
}
__device__ __forceinline__ float half_sum16(float x) {
#pragma unroll
    for (int m = 1; m < 16; m <<= 1) x += __shfl_xor(x, m, 32);
    return x;
}

__global__ __launch_bounds__(128) void flash_attn_kernel(const _Float16* __restrict__ q_h,
                                                         const _Float16* __restrict__ k_h,
                                                         const _Float16* __restrict__ v_t,
                                                         _Float16* __restrict__ o_h,
                                                         const int* __restrict__ causal_p) {
    constexpr int S = 2048, HD = 64;
    __shared__ _Float16 lds_p[4][16][32];   // per-wave P tile (C-layout -> A-layout bounce)

    const int wave = threadIdx.x >> 5;
    const int lane = threadIdx.x & 31;
    const int hi   = lane >> 4;
    const int l16  = lane & 15;
    const int h    = blockIdx.y;
    const int q0   = (blockIdx.x * 4 + wave) * 16;
    const int causal = *causal_p;
    const int koff = hi * 8;

    const _Float16* qrow = q_h + ((size_t)h * S + q0 + l16) * HD;
    F16x16 qa0, qa1;
    qa0.h[0] = *(const h8*)(qrow + koff);
    qa0.h[1] = *(const h8*)(qrow + koff + 16);
    qa1.h[0] = *(const h8*)(qrow + koff + 32);
    qa1.h[1] = *(const h8*)(qrow + koff + 48);

    float mrow[8], lrow[8];
#pragma unroll
    for (int i = 0; i < 8; ++i) { mrow[i] = -1.0e30f; lrow[i] = 0.0f; }
    v8f acc[4] = {};

    const int nkb = causal ? ((q0 + 16 + 31) / 32) : (S / 32);

    for (int kb = 0; kb < nkb; ++kb) {
        const int kb0 = kb * 32;

        // ---- scores: 16x32 tile = two 16x16 WMMA outputs, K-dim 64 each ----
        // Preload both K fragment pairs, then issue the 4 WMMAs with the two
        // accumulator chains interleaved for ILP.
        F16x16 kf0[2], kf1[2];
#pragma unroll
        for (int t = 0; t < 2; ++t) {
            const _Float16* krow = k_h + ((size_t)h * S + kb0 + t * 16 + l16) * HD;
            kf0[t].h[0] = *(const h8*)(krow + koff);
            kf0[t].h[1] = *(const h8*)(krow + koff + 16);
            kf1[t].h[0] = *(const h8*)(krow + koff + 32);
            kf1[t].h[1] = *(const h8*)(krow + koff + 48);
        }
        v8f sc[2] = {};
        sc[0] = WMMA_F16(qa0.v, kf0[0].v, sc[0]);
        sc[1] = WMMA_F16(qa0.v, kf0[1].v, sc[1]);
        sc[0] = WMMA_F16(qa1.v, kf1[0].v, sc[0]);
        sc[1] = WMMA_F16(qa1.v, kf1[1].v, sc[1]);

        // ---- scale + causal mask + online softmax ----
#pragma unroll
        for (int t = 0; t < 2; ++t) {
#pragma unroll
            for (int i = 0; i < 8; ++i) {
                float v = sc[t][i] * 0.125f;   // 1/sqrt(64)
                if (causal && (q0 + i + 8 * hi) < (kb0 + t * 16 + l16)) v = -1.0e30f;
                sc[t][i] = v;
            }
        }

#pragma unroll
        for (int i = 0; i < 8; ++i) {
            float rm = half_max16(fmaxf(sc[0][i], sc[1][i]));
            const float mnew = fmaxf(mrow[i], rm);
            const float corr = __expf(mrow[i] - mnew);
            const float p0 = __expf(sc[0][i] - mnew);
            const float p1 = __expf(sc[1][i] - mnew);
            const float rs = half_sum16(p0 + p1);
            lrow[i] = lrow[i] * corr + rs;
            mrow[i] = mnew;
#pragma unroll
            for (int t = 0; t < 4; ++t) acc[t][i] *= corr;
            const int row = i + 8 * hi;
            lds_p[wave][row][l16]      = (_Float16)p0;
            lds_p[wave][row][16 + l16] = (_Float16)p1;
        }
        __asm__ volatile("" ::: "memory");   // same-wave LDS ops are in-order

        // ---- P (A-layout from LDS) x V (contiguous-K from v_t) ----
        F16x16 pa;
        pa.h[0] = *(const h8*)&lds_p[wave][l16][koff];
        pa.h[1] = *(const h8*)&lds_p[wave][l16][koff + 16];
        F16x16 vb[4];
#pragma unroll
        for (int t = 0; t < 4; ++t) {
            const _Float16* vcol = v_t + ((size_t)h * HD + t * 16 + l16) * S + kb0;
            vb[t].h[0] = *(const h8*)(vcol + koff);
            vb[t].h[1] = *(const h8*)(vcol + koff + 16);
        }
#pragma unroll
        for (int t = 0; t < 4; ++t)
            acc[t] = WMMA_F16(pa.v, vb[t].v, acc[t]);
    }

#pragma unroll
    for (int t = 0; t < 4; ++t) {
#pragma unroll
        for (int i = 0; i < 8; ++i) {
            const int row = q0 + i + 8 * hi;
            const int col = h * HD + t * 16 + l16;
            o_h[(size_t)row * 1024 + col] = (_Float16)(acc[t][i] / lrow[i]);
        }
    }
}

// ---------------------------------------------------------------------------
// Host-side orchestration
// ---------------------------------------------------------------------------
extern "C" void kernel_launch(void* const* d_in, const int* in_sizes, int n_in,
                              void* d_out, int out_size, void* d_ws, size_t ws_size,
                              hipStream_t stream) {
    constexpr int S = 2048, D = 1024, H = 16, E = 3 * D;

    const float* x      = (const float*)d_in[0];
    const float* w_in   = (const float*)d_in[1];
    const float* w_out  = (const float*)d_in[2];
    const int*   causal = (const int*)d_in[3];
    float* out = (float*)d_out;

    char* ws = (char*)d_ws;
    _Float16* x_h     = (_Float16*)ws;  ws += (size_t)S * D * 2;
    _Float16* w_in_h  = (_Float16*)ws;  ws += (size_t)E * D * 2;
    _Float16* w_out_h = (_Float16*)ws;  ws += (size_t)D * D * 2;
    _Float16* q_h     = (_Float16*)ws;  ws += (size_t)S * D * 2;
    _Float16* k_h     = (_Float16*)ws;  ws += (size_t)S * D * 2;
    _Float16* v_t     = (_Float16*)ws;  ws += (size_t)S * D * 2;
    _Float16* o_h     = (_Float16*)ws;  ws += (size_t)S * D * 2;
    float*    qkv     = (float*)ws;     ws += (size_t)S * E * 4;

    cvt_f32_f16_kernel<<<(S * D) / 256, 256, 0, stream>>>(x, x_h, S * D);
    cvt_f32_f16_kernel<<<(E * D) / 256, 256, 0, stream>>>(w_in, w_in_h, E * D);
    cvt_f32_f16_kernel<<<(D * D) / 256, 256, 0, stream>>>(w_out, w_out_h, D * D);

    // qkv[S,3072] = x[S,1024] * w_in[3072,1024]^T
    gemm_rt_f16_kernel<<<dim3(E / 128, S / 64), 128, 0, stream>>>(x_h, w_in_h, qkv, S, E, D);

    // RoPE + head-major/transposed f16 layouts
    rope_kernel<<<(S * H * 32) / 256, 256, 0, stream>>>(qkv, q_h, k_h, v_t);

    // fused causal flash attention
    flash_attn_kernel<<<dim3(S / 64, H), 128, 0, stream>>>(q_h, k_h, v_t, o_h, causal);

    // out[S,1024] = o[S,1024] * w_out[1024,1024]^T
    gemm_rt_f16_kernel<<<dim3(D / 128, S / 64), 128, 0, stream>>>(o_h, w_out_h, out, S, D, D);
}
